// CrossAttention_1365799600776
// MI455X (gfx1250) — compile-verified, matrix operand order checked by reading, not run on hardware
//
#include <hip/hip_runtime.h>

// ---------------------------------------------------------------------------
// CDNA5 (gfx1250) cross-attention: bf16 WMMA GEMMs + flash-style attention.
// Wave32; V_WMMA_F32_16X16X32_BF16; async global->LDS staging (ASYNCcnt).
// ---------------------------------------------------------------------------

typedef __bf16 bf16;
typedef bf16 bf16x16 __attribute__((ext_vector_type(16)));
typedef float floatx8 __attribute__((ext_vector_type(8)));
typedef int intx4 __attribute__((ext_vector_type(4)));

#define DEV __device__ __forceinline__

// Problem constants
#define BATCH 8
#define SQ 4096
#define SKV 77
#define NKV 80   // SKV padded to multiple of 16 for QK^T N-dim
#define KPV 96   // SKV padded to multiple of 32 for PV K-dim
#define DE 1024
#define DC 768
#define NH 8
#define DH 128

// ---- async global->LDS (ASYNCcnt) support, guarded so compile never breaks
#if defined(__has_builtin)
#if __has_builtin(__builtin_amdgcn_global_load_async_to_lds_b128) && \
    __has_builtin(__builtin_amdgcn_s_wait_asynccnt)
#define USE_ASYNC 1
#endif
#endif
#ifndef USE_ASYNC
#define USE_ASYNC 0
#endif

#if USE_ASYNC
DEV void async_cp16(bf16* lds_dst, const bf16* gsrc) {
  // copy 16 bytes per lane, tracked by ASYNCcnt (global_load_async_to_lds_b128)
  // builtin signature: (int4 AS(1)* gsrc, int4 AS(3)* ldst, imm offset, imm cpol)
  __builtin_amdgcn_global_load_async_to_lds_b128(
      (__attribute__((address_space(1))) intx4*)gsrc,
      (__attribute__((address_space(3))) intx4*)lds_dst,
      /*offset=*/0, /*cpol=*/0);
}
DEV void async_wait0() { __builtin_amdgcn_s_wait_asynccnt(0); }
#else
DEV void async_wait0() {}
#endif

DEV floatx8 wmma_bf16(bf16x16 a, bf16x16 b, floatx8 c) {
  // D = A(16x32) * B(32x16) + C(16x16), f32 accumulate
  return __builtin_amdgcn_wmma_f32_16x16x32_bf16(
      /*neg_a=*/false, a, /*neg_b=*/false, b,
      /*c_mod=*/(short)0, c, /*reuse_a=*/false, /*reuse_b=*/false);
}

// Load one 16x32 bf16 operand fragment from LDS. ISA layout for both the
// A-matrix (16 rows x 32 K) and B-matrix (16 cols x 32 K) is identical:
//   lane = row/col (lane & 15); lanes 0-15 carry K = 0..7 and 16..23,
//   lanes 16-31 carry K = 8..15 and 24..31  (two 16-byte runs per lane).
DEV bf16x16 load_frag(const bf16* base, int ld) {
  const int lane = threadIdx.x & 31;
  const int r = lane & 15;
  const int kb = (lane & 16) ? 8 : 0;
  const bf16* p = base + r * ld + kb;
  bf16x16 f;
#pragma unroll
  for (int i = 0; i < 8; ++i) {
    f[i] = p[i];           // K = kb + 0..7   -> VGPR 0..3
    f[i + 8] = p[16 + i];  // K = kb + 16..23 -> VGPR 4..7
  }
  return f;
}

// ---------------------------------------------------------------------------
// Weight prep: fp32 W[K,N] -> bf16 W^T[N,K] (so B-tiles are 2D block copies)
// ---------------------------------------------------------------------------
__global__ void cvt_w_transpose(const float* __restrict__ src,
                                bf16* __restrict__ dst, int K, int N) {
  const int stride = gridDim.x * blockDim.x;
  const long total = (long)N * K;
  for (long i = blockIdx.x * blockDim.x + threadIdx.x; i < total; i += stride) {
    const long n = i / K, k = i - n * K;
    dst[i] = (bf16)src[k * N + n];
  }
}

// ---------------------------------------------------------------------------
// C[M,N] = A[M,K] @ W + bias, W given as bf16 W^T[N,K].
// Block tile 64x128, 4 waves (128 threads): each wave computes a 16x128 strip
// (8 WMMAs per 32-wide K chunk). A is converted fp32->bf16 while staging; the
// weight tile is staged with async global->LDS b128 chunks.
// ---------------------------------------------------------------------------
template <typename AT, typename OT>
__global__ void __launch_bounds__(128)
gemm_bias_kernel(const AT* __restrict__ A, const bf16* __restrict__ Wt,
                 const float* __restrict__ bias, OT* __restrict__ C,
                 int M, int N, int K) {
  constexpr int LDK = 40;  // 80-byte rows: keeps 16B chunks aligned, skews banks
  __shared__ __align__(16) bf16 As[64][LDK];    // A tile   [m][k]
  __shared__ __align__(16) bf16 Bs[128][LDK];   // W^T tile [n][k]

  const int n0 = blockIdx.x * 128;
  const int m0 = blockIdx.y * 64;
  const int tid = threadIdx.x;
  const int wave = tid >> 5;
  const int lane = tid & 31;

  floatx8 acc[8] = {};

  for (int k0 = 0; k0 < K; k0 += 32) {
    __syncthreads();
#if USE_ASYNC
    // Stage W^T (128 n x 32 k) as 512 x 16B async chunks (4 per thread).
#pragma unroll
    for (int c = tid; c < 512; c += 128) {
      const int n = c >> 2, koff = (c & 3) * 8;
      async_cp16(&Bs[n][koff], &Wt[(long)(n0 + n) * K + k0 + koff]);
    }
#else
    for (int i = tid; i < 128 * 32; i += 128) {
      const int n = i >> 5, k = i & 31;
      Bs[n][k] = Wt[(long)(n0 + n) * K + k0 + k];
    }
#endif
    // Stage A (64 m x 32 k), converting to bf16 (overlaps with async W copy).
    for (int i = tid; i < 64 * 32; i += 128) {
      const int r = i >> 5, k = i & 31;
      float v = (m0 + r < M) ? (float)A[(long)(m0 + r) * K + k0 + k] : 0.0f;
      As[r][k] = (bf16)v;
    }
    if (k0 + 32 < K)  // pull next A tile toward GL2 (global_prefetch_b8)
      __builtin_prefetch(&A[(long)(m0 + (tid >> 1)) * K + k0 + 32], 0, 1);
    async_wait0();
    __syncthreads();

    bf16x16 a = load_frag(&As[wave * 16][0], LDK);
#pragma unroll
    for (int j = 0; j < 8; ++j) {
      bf16x16 b = load_frag(&Bs[j * 16][0], LDK);
      acc[j] = wmma_bf16(a, b, acc[j]);
    }
  }

  // C/D layout: VGPR r holds (row = r + 8*(lane>>4), col = lane&15).
  const int half = lane >> 4, coll = lane & 15;
#pragma unroll
  for (int j = 0; j < 8; ++j) {
    const float bv = bias[n0 + j * 16 + coll];
#pragma unroll
    for (int r = 0; r < 8; ++r) {
      const int row = m0 + wave * 16 + r + half * 8;
      if (row < M)
        C[(long)row * N + n0 + j * 16 + coll] = (OT)(acc[j][r] + bv);
    }
  }
}

// ---------------------------------------------------------------------------
// Attention: one block per (b, h, 64-row q tile). 4 waves, 16 q rows each.
// S = Q K^T (K padded to 80 cols), softmax in-register, O = P V (K padded 96).
// Q/K/V are bf16 in [B, S, 1024] layout (heads are 128-col slices).
// ---------------------------------------------------------------------------
__global__ void __launch_bounds__(128)
attn_kernel(const bf16* __restrict__ Q, const bf16* __restrict__ Kb,
            const bf16* __restrict__ Vb, bf16* __restrict__ O) {
  __shared__ __align__(16) bf16 Qs[64][DH];      // 16 KB  [q][d]
  __shared__ __align__(16) bf16 Ks[NKV][DH];     // 20 KB  [token][d]
  __shared__ __align__(16) bf16 Vs[DH][KPV];     // 24 KB  [d][token]
  __shared__ __align__(16) bf16 Ps[4][16][KPV];  // 12 KB  per-wave P

  const int b = blockIdx.z, h = blockIdx.y;
  const int q0 = blockIdx.x * 64;
  const int tid = threadIdx.x;
  const int wave = tid >> 5;
  const int lane = tid & 31;

  // ---- stage Q and K tiles (pure copies -> async b128 chunks) ----
#if USE_ASYNC
#pragma unroll
  for (int c = tid; c < 64 * (DH / 8); c += 128) {  // 1024 chunks
    const int r = c >> 4, off = (c & 15) * 8;
    async_cp16(&Qs[r][off], &Q[((long)b * SQ + q0 + r) * DE + h * DH + off]);
  }
  for (int c = tid; c < SKV * (DH / 8); c += 128) {  // 1232 chunks
    const int t = c >> 4, off = (c & 15) * 8;
    async_cp16(&Ks[t][off], &Kb[((long)b * SKV + t) * DE + h * DH + off]);
  }
#else
  for (int i = tid; i < 64 * DH; i += 128) {
    const int r = i >> 7, d = i & 127;
    Qs[r][d] = Q[((long)b * SQ + q0 + r) * DE + h * DH + d];
  }
  for (int i = tid; i < SKV * DH; i += 128) {
    const int t = i >> 7, d = i & 127;
    Ks[t][d] = Kb[((long)b * SKV + t) * DE + h * DH + d];
  }
#endif
  // zero-pad K token rows 77..79
  for (int i = tid; i < 3 * DH; i += 128)
    Ks[SKV + (i >> 7)][i & 127] = (bf16)0.0f;
  // V staged transposed [d][t] (B-frag layout for PV); pad tokens with zero
  for (int i = tid; i < DH * KPV; i += 128) {
    const int d = i / KPV, t = i % KPV;
    Vs[d][t] = (t < SKV) ? Vb[((long)b * SKV + t) * DE + h * DH + d]
                         : (bf16)0.0f;
  }
  async_wait0();
  __syncthreads();

  // ---- S = Q K^T : 16 q rows x 80 tokens, f32 accum ----
  floatx8 s[5] = {};
#pragma unroll
  for (int kc = 0; kc < DH / 32; ++kc) {
    bf16x16 aq = load_frag(&Qs[wave * 16][kc * 32], DH);
#pragma unroll
    for (int j = 0; j < 5; ++j) {
      bf16x16 bk = load_frag(&Ks[j * 16][kc * 32], DH);
      s[j] = wmma_bf16(aq, bk, s[j]);
    }
  }

  // ---- softmax over tokens (row-wise). Row lives in one 16-lane half. ----
  const float scale = 0.08838834764831845f;  // 1/sqrt(128)
  const int half = lane >> 4, coll = lane & 15;
#pragma unroll
  for (int r = 0; r < 8; ++r) {
    float m = -1e30f;
#pragma unroll
    for (int j = 0; j < 5; ++j) {
      float v = s[j][r] * scale;
      if (j * 16 + coll >= SKV) v = -1e30f;  // mask padding tokens
      s[j][r] = v;
      m = fmaxf(m, v);
    }
#pragma unroll
    for (int o = 8; o >= 1; o >>= 1) m = fmaxf(m, __shfl_xor(m, o, 32));
    float sum = 0.0f;
#pragma unroll
    for (int j = 0; j < 5; ++j) {
      const float e = __expf(s[j][r] - m);  // masked -> exp(-huge) == 0
      s[j][r] = e;
      sum += e;
    }
#pragma unroll
    for (int o = 8; o >= 1; o >>= 1) sum += __shfl_xor(sum, o, 32);
    const float inv = 1.0f / sum;
    const int row = r + half * 8;
#pragma unroll
    for (int j = 0; j < 5; ++j)
      Ps[wave][row][j * 16 + coll] = (bf16)(s[j][r] * inv);
  }
  // zero P padding columns 80..95 (K-dim pad for PV)
  for (int i = lane; i < 16 * 16; i += 32)
    Ps[wave][i >> 4][NKV + (i & 15)] = (bf16)0.0f;
  __syncthreads();

  // ---- O = P V : 16 q rows x 128 d, K = 96 (3 chunks of 32) ----
  floatx8 o[8] = {};
#pragma unroll
  for (int kc = 0; kc < KPV / 32; ++kc) {
    bf16x16 ap = load_frag(&Ps[wave][0][kc * 32], KPV);
#pragma unroll
    for (int j = 0; j < 8; ++j) {
      bf16x16 bv = load_frag(&Vs[j * 16][kc * 32], KPV);
      o[j] = wmma_bf16(ap, bv, o[j]);
    }
  }

  // ---- store merged-head attention output, bf16 [B, SQ, 1024] ----
#pragma unroll
  for (int j = 0; j < 8; ++j)
#pragma unroll
    for (int r = 0; r < 8; ++r) {
      const int row = q0 + wave * 16 + r + half * 8;
      const int col = h * DH + j * 16 + coll;
      O[((long)b * SQ + row) * DE + col] = (bf16)o[j][r];
    }
}

// ---------------------------------------------------------------------------
// Launch: weight transpose+convert -> Q/K/V proj -> attention -> out proj.
// ---------------------------------------------------------------------------
extern "C" void kernel_launch(void* const* d_in, const int* in_sizes, int n_in,
                              void* d_out, int out_size, void* d_ws,
                              size_t ws_size, hipStream_t stream) {
  (void)in_sizes; (void)n_in; (void)out_size; (void)ws_size;
  const float* x  = (const float*)d_in[0];
  const float* y  = (const float*)d_in[1];
  const float* Wq = (const float*)d_in[2];
  const float* bq = (const float*)d_in[3];
  const float* Wk = (const float*)d_in[4];
  const float* bk = (const float*)d_in[5];
  const float* Wv = (const float*)d_in[6];
  const float* bv = (const float*)d_in[7];
  const float* Wo = (const float*)d_in[8];
  const float* bo = (const float*)d_in[9];
  float* out = (float*)d_out;

  // Workspace carve-out (bf16), 256B aligned. Total ~138 MB.
  char* ws = (char*)d_ws;
  size_t off = 0;
  auto take = [&](size_t elems) -> bf16* {
    bf16* p = (bf16*)(ws + off);
    off += (elems * sizeof(bf16) + 255) & ~(size_t)255;
    return p;
  };
  bf16* wqb = take((size_t)DE * DE);   // W^T layouts, [N][K]
  bf16* wkb = take((size_t)DE * DC);
  bf16* wvb = take((size_t)DE * DC);
  bf16* wob = take((size_t)DE * DE);
  bf16* Qb  = take((size_t)BATCH * SQ * DE);
  bf16* Kbf = take((size_t)BATCH * SKV * DE);
  bf16* Vbf = take((size_t)BATCH * SKV * DE);
  bf16* Ob  = take((size_t)BATCH * SQ * DE);

  const dim3 cvtG(256), cvtB(256);
  cvt_w_transpose<<<cvtG, cvtB, 0, stream>>>(Wq, wqb, DE, DE);
  cvt_w_transpose<<<cvtG, cvtB, 0, stream>>>(Wk, wkb, DC, DE);
  cvt_w_transpose<<<cvtG, cvtB, 0, stream>>>(Wv, wvb, DC, DE);
  cvt_w_transpose<<<cvtG, cvtB, 0, stream>>>(Wo, wob, DE, DE);

  const int MQ = BATCH * SQ;    // 32768
  const int MKV = BATCH * SKV;  // 616
  const dim3 blk(128);

  // Q = x @ Wq + bq  -> bf16
  gemm_bias_kernel<float, bf16>
      <<<dim3(DE / 128, (MQ + 63) / 64), blk, 0, stream>>>(
          x, wqb, bq, Qb, MQ, DE, DE);
  // K = y @ Wk + bk  -> bf16
  gemm_bias_kernel<float, bf16>
      <<<dim3(DE / 128, (MKV + 63) / 64), blk, 0, stream>>>(
          y, wkb, bk, Kbf, MKV, DE, DC);
  // V = y @ Wv + bv  -> bf16
  gemm_bias_kernel<float, bf16>
      <<<dim3(DE / 128, (MKV + 63) / 64), blk, 0, stream>>>(
          y, wvb, bv, Vbf, MKV, DE, DC);

  // attention per (b, h, 64-row q tile)
  attn_kernel<<<dim3(SQ / 64, NH, BATCH), blk, 0, stream>>>(Qb, Kbf, Vbf, Ob);

  // out = attn @ Wo + bo -> fp32
  gemm_bias_kernel<bf16, float>
      <<<dim3(DE / 128, (MQ + 63) / 64), blk, 0, stream>>>(
          Ob, wob, bo, out, MQ, DE, DE);
}
